// CausalSelfAttention_49761491091745
// MI455X (gfx1250) — matrix operand order repaired and checked by reading, not compiled, not run on hardware
//
#include <hip/hip_runtime.h>
#include <stdint.h>

// ---------------------------------------------------------------------------
// CDNA5 (gfx1250) causal self-attention: bf16 WMMA everywhere, f32 softmax.
// B=4, T=2048, C=1024, H=16, D=64.  M = B*T = 8192.
// GEMMs use 64x64 register blocking per wave: 16 WMMAs per 8 fragment loads.
// ---------------------------------------------------------------------------

typedef __bf16 bf16_t;
typedef __attribute__((ext_vector_type(16))) __bf16   v16bf;
typedef __attribute__((ext_vector_type(8)))  float    v8f;
typedef __attribute__((ext_vector_type(8)))  uint32_t v8u;

#define WMMA_BF16(a, b, c) \
  __builtin_amdgcn_wmma_f32_16x16x32_bf16(false, (a), false, (b), (short)0, (c), false, false)

static __device__ __forceinline__ bf16_t f2bf(float f) {
  union { float f; uint32_t u; } x; x.f = f;
  uint32_t u = x.u;
  uint32_t r = (u + 0x7FFFu + ((u >> 16) & 1u)) >> 16;   // round-to-nearest-even
  unsigned short hs = (unsigned short)r;
  bf16_t o; __builtin_memcpy(&o, &hs, 2);
  return o;
}

// A-matrix fragment, 16x32 bf16, row-major source (stride in elements).
// ISA layout: lane L -> row M = L%16; VGPR j holds K = (j>=4?16:0) + (L/16)*8 + 2*(j%4) .. +1
static __device__ __forceinline__ v16bf load_a16x32(const bf16_t* base, int stride, int lane) {
  const int half = (lane >> 4) & 1;
  const bf16_t* p = base + (size_t)(lane & 15) * stride;
  v8u u;
#pragma unroll
  for (int j = 0; j < 8; ++j) {
    const int k = ((j & 4) ? 16 : 0) + (half << 3) + ((j & 3) << 1);
    u[j] = *(const uint32_t*)(p + k);
  }
  return __builtin_bit_cast(v16bf, u);
}

// B-matrix fragment, 32x16 bf16, source laid out K-contiguous per column
// (i.e. element (k,n) at base[n*stride + k]).
// ISA layout: lane L -> col N = L%16; VGPR j holds K = (j>=4?16:0) + (L/16)*8 + 2*(j%4) .. +1
static __device__ __forceinline__ v16bf load_b32x16(const bf16_t* base, int stride, int lane) {
  const int half = (lane >> 4) & 1;
  const bf16_t* p = base + (size_t)(lane & 15) * stride;
  v8u u;
#pragma unroll
  for (int j = 0; j < 8; ++j) u[j] = *(const uint32_t*)(p + ((j & 4) ? 16 : 0) + (half << 3) + ((j & 3) << 1));
  return __builtin_bit_cast(v16bf, u);
}

// ---------------------------------------------------------------------------
// Elementwise conversions
// ---------------------------------------------------------------------------
__global__ void cvt_f32_bf16(const float* __restrict__ in, bf16_t* __restrict__ out, int n) {
  int i = blockIdx.x * blockDim.x + threadIdx.x;
  if (i < n) out[i] = f2bf(in[i]);
}

// w[K][N] f32  ->  wT[N][K] bf16
__global__ void transpose_to_bf16(const float* __restrict__ w, bf16_t* __restrict__ wT,
                                  int K, int N) {
  int i = blockIdx.x * blockDim.x + threadIdx.x;
  if (i < N * K) {
    int n = i / K;
    int k = i - n * K;
    wT[i] = f2bf(w[(size_t)k * N + n]);
  }
}

// ---------------------------------------------------------------------------
// 64x64-per-wave GEMM core over A[M x K] (row-major) and BT[N x K]
// (K-contiguous per output column).  acc[i][j] = 16x16 tile (m0+16i, n0+16j).
// Per K-step: 4 A frags + 4 B frags (8 x b128-clause loads) feed 16 WMMAs.
// ---------------------------------------------------------------------------
static __device__ __forceinline__ void gemm64_core(const bf16_t* __restrict__ A,
                                                   const bf16_t* __restrict__ BT,
                                                   int m0, int n0, int Kdim, int lane,
                                                   v8f acc[4][4]) {
  const bf16_t* ap = A  + (size_t)m0 * Kdim;
  const bf16_t* bp = BT + (size_t)n0 * Kdim;
  for (int kk = 0; kk < Kdim; kk += 32) {
    __builtin_prefetch(ap + kk + 256, 0, 0);  // global_prefetch_b8
    v16bf a[4], b[4];
#pragma unroll
    for (int i = 0; i < 4; ++i)
      a[i] = load_a16x32(ap + (size_t)(16 * i) * Kdim + kk, Kdim, lane);
#pragma unroll
    for (int j = 0; j < 4; ++j)
      b[j] = load_b32x16(bp + (size_t)(16 * j) * Kdim + kk, Kdim, lane);
#pragma unroll
    for (int i = 0; i < 4; ++i)
#pragma unroll
      for (int j = 0; j < 4; ++j)
        acc[i][j] = WMMA_BF16(a[i], b[j], acc[i][j]);
  }
}

// ---------------------------------------------------------------------------
// QKV GEMM: [8192 x 1024] @ [1024 x 3072] -> scatter Q,K [BH][T][D], V^T [BH][D][T]
// 128 m-tiles * 48 n-tiles of 64x64 = 6144 waves.
// The whole 64x64 tile maps to one (which, head, batch): n0 is 64-aligned so
// which/h are wave-uniform; m0 is 64-aligned within a 2048-row batch so b is
// uniform.  Epilogue is a single uniform select + affine stores (no per-element
// branching).
// ---------------------------------------------------------------------------
__global__ __launch_bounds__(256) void qkv_gemm(const bf16_t* __restrict__ X,
                                                const bf16_t* __restrict__ WT,
                                                bf16_t* __restrict__ Q,
                                                bf16_t* __restrict__ Kc,
                                                bf16_t* __restrict__ VT) {
  const int lane = threadIdx.x & 31;
  const int half = lane >> 4;
  const int wid  = blockIdx.x * (blockDim.x >> 5) + (threadIdx.x >> 5);
  const int NT   = 3072 / 64;                 // 48 n-tiles of 64
  const int m0   = (wid / NT) << 6;
  const int n0   = (wid % NT) << 6;

  v8f acc[4][4] = {};
  gemm64_core(X, WT, m0, n0, 1024, lane, acc);

  // wave-uniform tile coordinates
  const int which = n0 >> 10;                 // 0=q 1=k 2=v
  const int h     = (n0 & 1023) >> 6;         // head (tile spans one 64-chunk)
  const int b_    = m0 >> 11;                 // batch (tile within one batch)
  const int t0    = m0 & 2047;                // first row's t
  const size_t bh = (size_t)(b_ * 16 + h);

  if (which == 2) {
    // V^T [BH][D][T]: element (d, t) at bh*64*2048 + d*2048 + t
    bf16_t* vt = VT + bh * 64 * 2048 + t0;
#pragma unroll
    for (int j = 0; j < 4; ++j) {
      const int d = j * 16 + (lane & 15);
#pragma unroll
      for (int i = 0; i < 4; ++i)
#pragma unroll
        for (int r = 0; r < 8; ++r)
          vt[(size_t)d * 2048 + i * 16 + r + (half << 3)] = f2bf(acc[i][j][r]);
    }
  } else {
    // Q/K [BH][T][D]: element (t, d) at (bh*2048 + t)*64 + d
    bf16_t* qk = (which == 0 ? Q : Kc) + (bh * 2048 + t0) * 64;
#pragma unroll
    for (int j = 0; j < 4; ++j) {
      const int d = j * 16 + (lane & 15);
#pragma unroll
      for (int i = 0; i < 4; ++i)
#pragma unroll
        for (int r = 0; r < 8; ++r)
          qk[(size_t)(i * 16 + r + (half << 3)) * 64 + d] = f2bf(acc[i][j][r]);
    }
  }
}

// ---------------------------------------------------------------------------
// Flash attention: 4 waves/block, 16 q-rows/wave, key blocks of 32.
// ---------------------------------------------------------------------------
__global__ __launch_bounds__(128) void attn(const bf16_t* __restrict__ Q,
                                            const bf16_t* __restrict__ Kc,
                                            const bf16_t* __restrict__ VT,
                                            bf16_t* __restrict__ CTX) {
  const int lane  = threadIdx.x & 31;
  const int half  = lane >> 4;
  const int wave  = threadIdx.x >> 5;
  const int bh    = blockIdx.x >> 5;          // B*H = 64
  const int qblk  = blockIdx.x & 31;          // T/64 = 32 blocks of 64 rows
  const int qbase = (qblk << 6) + (wave << 4);
  const float scale = 0.125f;                 // 1/sqrt(64)

  __shared__ __align__(16) bf16_t lds[4][16][36];   // per-wave P tile, padded stride

  const bf16_t* qp = Q + ((size_t)bh * 2048 + qbase) * 64;
  const v16bf aq0 = load_a16x32(qp,      64, lane); // d = 0..31
  const v16bf aq1 = load_a16x32(qp + 32, 64, lane); // d = 32..63

  float m_r[8], l_r[8];
#pragma unroll
  for (int r = 0; r < 8; ++r) { m_r[r] = -1e30f; l_r[r] = 0.f; }
  v8f acc[4] = {};
  const v8f zero = {};

  const bf16_t* kp0 = Kc + (size_t)bh * 2048 * 64;
  const bf16_t* vp0 = VT + (size_t)bh * 64 * 2048;

  for (int kb = 0; kb < qbase + 16; kb += 32) {
    // --- scores: S[16q x 32k] = Q[16x64] @ K^T[64x32], scaled + causal mask ---
    v8f s[2];
#pragma unroll
    for (int nt = 0; nt < 2; ++nt) {
      const bf16_t* kp = kp0 + (size_t)(kb + nt * 16) * 64;
      v16bf b0 = load_b32x16(kp,      64, lane);    // d = 0..31
      v16bf b1 = load_b32x16(kp + 32, 64, lane);    // d = 32..63
      v8f sc = WMMA_BF16(aq0, b0, zero);
      sc = WMMA_BF16(aq1, b1, sc);
      const int tk = kb + nt * 16 + (lane & 15);
#pragma unroll
      for (int r = 0; r < 8; ++r) {
        const int tq = qbase + r + (half << 3);
        s[nt][r] = (tk <= tq) ? sc[r] * scale : -1e30f;
      }
    }

    // --- online softmax (rows live in 16-lane halves at fixed VGPR index) ---
    float alpha[8];
#pragma unroll
    for (int r = 0; r < 8; ++r) {
      float mx = fmaxf(s[0][r], s[1][r]);
#pragma unroll
      for (int off = 1; off < 16; off <<= 1) mx = fmaxf(mx, __shfl_xor(mx, off, 32));
      const float mn = fmaxf(m_r[r], mx);
      alpha[r] = __expf(m_r[r] - mn);
      const float p0 = __expf(s[0][r] - mn);
      const float p1 = __expf(s[1][r] - mn);
      s[0][r] = p0; s[1][r] = p1;
      float ps = p0 + p1;
#pragma unroll
      for (int off = 1; off < 16; off <<= 1) ps += __shfl_xor(ps, off, 32);
      l_r[r] = l_r[r] * alpha[r] + ps;
      m_r[r] = mn;
    }
#pragma unroll
    for (int j = 0; j < 4; ++j)
#pragma unroll
      for (int r = 0; r < 8; ++r) acc[j][r] = acc[j][r] * alpha[r];

    // --- re-layout P (C/D layout) -> A-fragment via per-wave LDS tile ---
    // LDS ops from one wave complete in order (DScnt), so no barrier needed.
#pragma unroll
    for (int r = 0; r < 8; ++r) {
      const int M = r + (half << 3);
      lds[wave][M][lane & 15]        = f2bf(s[0][r]);
      lds[wave][M][16 + (lane & 15)] = f2bf(s[1][r]);
    }
    const v16bf pa = load_a16x32(&lds[wave][0][0], 36, lane);

    // --- ctx += P[16x32] @ V[32x64]  (V^T gives K-contiguous B fragments) ---
#pragma unroll
    for (int j = 0; j < 4; ++j) {
      v16bf bv = load_b32x16(vp0 + (size_t)(j * 16) * 2048 + kb, 2048, lane);
      acc[j] = WMMA_BF16(pa, bv, acc[j]);
    }
  }

  // --- normalize and store ctx (bf16, [B][T][C]) ---
  const int b_ = bh >> 4;
  const int h  = bh & 15;
  bf16_t* cp = CTX + (size_t)(b_ * 2048 + qbase) * 1024 + h * 64;
#pragma unroll
  for (int r = 0; r < 8; ++r) {
    const float inv = 1.f / l_r[r];
    const int trow = r + (half << 3);
#pragma unroll
    for (int j = 0; j < 4; ++j)
      cp[(size_t)trow * 1024 + j * 16 + (lane & 15)] = f2bf(acc[j][r] * inv);
  }
}

// ---------------------------------------------------------------------------
// Output projection: ctx[8192 x 1024] @ w_out[1024 x 1024] -> f32 out
// 128 m-tiles * 16 n-tiles of 64x64 = 2048 waves.
// ---------------------------------------------------------------------------
__global__ __launch_bounds__(256) void out_gemm(const bf16_t* __restrict__ A,
                                                const bf16_t* __restrict__ WT,
                                                float* __restrict__ out) {
  const int lane = threadIdx.x & 31;
  const int half = lane >> 4;
  const int wid  = blockIdx.x * (blockDim.x >> 5) + (threadIdx.x >> 5);
  const int NT   = 1024 / 64;                 // 16 n-tiles of 64
  const int m0   = (wid / NT) << 6;
  const int n0   = (wid % NT) << 6;

  v8f acc[4][4] = {};
  gemm64_core(A, WT, m0, n0, 1024, lane, acc);

  float* op = out + (size_t)m0 * 1024 + n0;
#pragma unroll
  for (int j = 0; j < 4; ++j) {
    const int col = j * 16 + (lane & 15);
#pragma unroll
    for (int i = 0; i < 4; ++i)
#pragma unroll
      for (int r = 0; r < 8; ++r)
        op[(size_t)(i * 16 + r + (half << 3)) * 1024 + col] = acc[i][j][r];
  }
}

// ---------------------------------------------------------------------------
extern "C" void kernel_launch(void* const* d_in, const int* in_sizes, int n_in,
                              void* d_out, int out_size, void* d_ws, size_t ws_size,
                              hipStream_t stream) {
  const float* x     = (const float*)d_in[0];   // [4,2048,1024]
  const float* w_qkv = (const float*)d_in[1];   // [1024,3072]
  const float* w_out = (const float*)d_in[2];   // [1024,1024]
  float* out = (float*)d_out;                   // [4,2048,1024]

  char* ws = (char*)d_ws;
  size_t off = 0;
  auto alloc = [&](size_t bytes) -> void* {
    void* p = ws + off;
    off += (bytes + 255) & ~(size_t)255;
    return p;
  };

  bf16_t* xb   = (bf16_t*)alloc((size_t)8192 * 1024 * 2);    // x bf16
  bf16_t* wqT  = (bf16_t*)alloc((size_t)3072 * 1024 * 2);    // w_qkv^T bf16
  bf16_t* woT  = (bf16_t*)alloc((size_t)1024 * 1024 * 2);    // w_out^T bf16
  bf16_t* Qb   = (bf16_t*)alloc((size_t)64 * 2048 * 64 * 2); // [BH][T][D]
  bf16_t* Kb   = (bf16_t*)alloc((size_t)64 * 2048 * 64 * 2); // [BH][T][D]
  bf16_t* VTb  = (bf16_t*)alloc((size_t)64 * 64 * 2048 * 2); // [BH][D][T]
  bf16_t* ctxb = (bf16_t*)alloc((size_t)8192 * 1024 * 2);    // [B][T][C]
  (void)ws_size; (void)in_sizes; (void)n_in; (void)out_size;

  const int nx = 8192 * 1024;
  cvt_f32_bf16<<<(nx + 255) / 256, 256, 0, stream>>>(x, xb, nx);

  const int nwq = 1024 * 3072;
  transpose_to_bf16<<<(nwq + 255) / 256, 256, 0, stream>>>(w_qkv, wqT, 1024, 3072);
  const int nwo = 1024 * 1024;
  transpose_to_bf16<<<(nwo + 255) / 256, 256, 0, stream>>>(w_out, woT, 1024, 1024);

  // 128 m-tiles * 48 n-tiles = 6144 waves, 8 waves/block -> 768 blocks
  qkv_gemm<<<768, 256, 0, stream>>>(xb, wqT, Qb, Kb, VTb);

  // B*H * (T/64) = 64*32 = 2048 blocks of 4 waves
  attn<<<64 * 32, 128, 0, stream>>>(Qb, Kb, VTb, ctxb);

  // 128 m-tiles * 16 n-tiles = 2048 waves, 8 waves/block -> 256 blocks
  out_gemm<<<256, 256, 0, stream>>>(ctxb, woT, out);
}
